// GNNlayer_85529978732561
// MI455X (gfx1250) — matrix operand order repaired and checked by reading, not compiled, not run on hardware
//
#include <hip/hip_runtime.h>

// Problem constants (from reference)
#define NNODES 50000
#define NEDGE  150000
#define ETOT   (NEDGE + NNODES)   // self-loops appended
#define DH_    128
#define NHEAD  8
#define HDIM   (NHEAD * DH_)      // 1024
#define VOCAB  32000
#define TTARG  1024

typedef __attribute__((ext_vector_type(16))) __bf16       bf16x16;
typedef __attribute__((ext_vector_type(8)))  float        f32x8;
typedef __attribute__((ext_vector_type(4)))  unsigned int u32x4;

union V16 { u32x4 u[2]; bf16x16 v; };

__device__ __forceinline__ unsigned short f2bf(float f) {
    unsigned u = __float_as_uint(f);
    u += 0x7fffu + ((u >> 16) & 1u);          // round-to-nearest-even
    return (unsigned short)(u >> 16);
}
__device__ __forceinline__ float bf2f(unsigned short h) {
    return __uint_as_float(((unsigned)h) << 16);
}
// order-preserving float<->uint map for atomicMax on floats
__device__ __forceinline__ unsigned fenc(float f) {
    unsigned u = __float_as_uint(f);
    return (u & 0x80000000u) ? ~u : (u | 0x80000000u);
}
__device__ __forceinline__ float fdec(unsigned k) {
    unsigned u = (k & 0x80000000u) ? (k & 0x7fffffffu) : ~k;
    return __uint_as_float(u);
}
__device__ __forceinline__ void edge_sd(const int* __restrict__ ei, int e, int& s, int& d) {
    if (e < NEDGE) { s = ei[e]; d = ei[NEDGE + e]; }
    else           { s = d = e - NEDGE; }      // self-loop
}

// ---------------------------------------------------------------------------
// Weight transpose + fp32->bf16:  Wt[n][k] = bf16(W[k][n])
__global__ void transpose_bf16_kernel(const float* __restrict__ W,
                                      unsigned short* __restrict__ Wt,
                                      int K, int N) {
    int idx = blockIdx.x * blockDim.x + threadIdx.x;
    if (idx >= K * N) return;
    int n = idx / K, k = idx - n * K;
    Wt[(size_t)n * K + k] = f2bf(W[(size_t)k * N + n]);
}

// ---------------------------------------------------------------------------
// x = layernorm(emb[y]) -> bf16, one block (128 threads) per node
__global__ void embed_ln_kernel(const int* __restrict__ y,
                                const float* __restrict__ emb,
                                unsigned short* __restrict__ xbf) {
    __shared__ float red[128];
    const int i = blockIdx.x, c = threadIdx.x;
    float v = emb[(size_t)y[i] * DH_ + c];
    red[c] = v; __syncthreads();
    for (int s = 64; s > 0; s >>= 1) { if (c < s) red[c] += red[c + s]; __syncthreads(); }
    float mean = red[0] * (1.0f / DH_); __syncthreads();
    float d = v - mean;
    red[c] = d * d; __syncthreads();
    for (int s = 64; s > 0; s >>= 1) { if (c < s) red[c] += red[c + s]; __syncthreads(); }
    float var = red[0] * (1.0f / DH_);
    xbf[(size_t)i * DH_ + c] = f2bf(d * rsqrtf(var + 1e-5f));
}

// ---------------------------------------------------------------------------
// bf16 WMMA GEMM: C[M,N] = act(A[M,K] @ Bt[N,K]^T + bias)
// one wave per 16x16 tile, K stepped by 32 via v_wmma_f32_16x16x32_bf16.
// act: 0 = none, 2 = exact GELU. Writes bf16 (Cbf) or fp32 (Cf).
__global__ __launch_bounds__(256)
void wmma_gemm_kernel(const unsigned short* __restrict__ A,
                      const unsigned short* __restrict__ Bt,
                      const float* __restrict__ bias,
                      float* __restrict__ Cf,
                      unsigned short* __restrict__ Cbf,
                      int M, int N, int K, int act) {
    const int lane = threadIdx.x & 31;
    const int wave = threadIdx.x >> 5;
    const int tile = blockIdx.x * (blockDim.x >> 5) + wave;
    const int tiles_n = N >> 4;
    if (tile >= (M >> 4) * tiles_n) return;          // wave-uniform exit
    const int tm = tile / tiles_n;
    const int tn = tile - tm * tiles_n;
    const int rl = lane & 15, hi = lane >> 4;

    const unsigned short* arow = A  + (size_t)(tm * 16 + rl) * K;
    const unsigned short* brow = Bt + (size_t)(tn * 16 + rl) * K;

    f32x8 acc = {};
    for (int kk = 0; kk < K; kk += 32) {
        V16 a, b;
        // A 16x32 bf16 lane layout: lanes0-15 K 0-7 / 16-23, lanes16-31 K 8-15 / 24-31
        a.u[0] = *(const u32x4*)(arow + kk + hi * 8);
        a.u[1] = *(const u32x4*)(arow + kk + 16 + hi * 8);
        // B 32x16 lane layout: lanes0-15 K 0-7 / 16-23, lanes16-31 K 8-15 / 24-31 (col = lane&15)
        b.u[0] = *(const u32x4*)(brow + kk + hi * 8);
        b.u[1] = *(const u32x4*)(brow + kk + 16 + hi * 8);
        acc = __builtin_amdgcn_wmma_f32_16x16x32_bf16(false, a.v, false, b.v,
                                                      (short)0, acc, false, false);
    }

    const int col = tn * 16 + rl;
    const float bv = bias ? bias[col] : 0.0f;
#pragma unroll
    for (int j = 0; j < 8; ++j) {                    // C layout: M = hi*8 + j
        int row = tm * 16 + hi * 8 + j;
        float v = acc[j] + bv;
        if (act == 2) v = 0.5f * v * (1.0f + erff(v * 0.70710678f));
        size_t off = (size_t)row * N + col;
        if (Cbf) Cbf[off] = f2bf(v); else Cf[off] = v;
    }
}

// ---------------------------------------------------------------------------
// per-(node,head) attention terms: asn = <h[i,h,:], a_src[h,:]>, adn likewise
__global__ void attn_score_kernel(const unsigned short* __restrict__ hb,
                                  const float* __restrict__ a_src,
                                  const float* __restrict__ a_dst,
                                  float* __restrict__ asn,
                                  float* __restrict__ adn) {
    int idx = blockIdx.x * blockDim.x + threadIdx.x;
    if (idx >= NNODES * NHEAD) return;
    int i = idx >> 3, h = idx & 7;
    const unsigned short* hp = hb + (size_t)i * HDIM + h * DH_;
    const float* as = a_src + h * DH_;
    const float* ad = a_dst + h * DH_;
    float s0 = 0.f, s1 = 0.f;
    for (int c = 0; c < DH_; ++c) {
        float v = bf2f(hp[c]);
        s0 += v * as[c]; s1 += v * ad[c];
    }
    asn[idx] = s0; adn[idx] = s1;
}

// segment-max pass (alpha = leaky_relu(as[src]+ad[dst], 0.2))
__global__ void edge_max_kernel(const int* __restrict__ ei,
                                const float* __restrict__ asn,
                                const float* __restrict__ adn,
                                float* __restrict__ alpha,
                                unsigned* __restrict__ amax) {
    int idx = blockIdx.x * blockDim.x + threadIdx.x;
    if (idx >= ETOT * NHEAD) return;
    int e = idx >> 3, h = idx & 7, s, d;
    edge_sd(ei, e, s, d);
    float a = asn[s * NHEAD + h] + adn[d * NHEAD + h];
    a = a > 0.f ? a : 0.2f * a;
    alpha[idx] = a;
    atomicMax(&amax[d * NHEAD + h], fenc(a));
}

// exp + segment-sum pass
__global__ void edge_exp_kernel(const int* __restrict__ ei,
                                float* __restrict__ alpha,
                                const unsigned* __restrict__ amax,
                                float* __restrict__ denom) {
    int idx = blockIdx.x * blockDim.x + threadIdx.x;
    if (idx >= ETOT * NHEAD) return;
    int e = idx >> 3, h = idx & 7, s, d;
    edge_sd(ei, e, s, d);
    float ex = __expf(alpha[idx] - fdec(amax[d * NHEAD + h]));
    alpha[idx] = ex;
    atomicAdd(&denom[d * NHEAD + h], ex);
}

// message aggregation: agg[dst,h,c] += h[src,h,c] * coef  (one block per edge)
__global__ void edge_agg_kernel(const int* __restrict__ ei,
                                const unsigned short* __restrict__ hb,
                                const float* __restrict__ exb,
                                const float* __restrict__ denom,
                                float* __restrict__ agg) {
    __shared__ float coef[NHEAD];
    const int e = blockIdx.x, c = threadIdx.x;
    int s, d;
    edge_sd(ei, e, s, d);
    if (c < NHEAD) coef[c] = exb[e * NHEAD + c] / denom[d * NHEAD + c];
    __syncthreads();
    const unsigned short* hp = hb + (size_t)s * HDIM + c;
    float* op = agg + (size_t)d * HDIM + c;
#pragma unroll
    for (int h = 0; h < NHEAD; ++h)
        atomicAdd(op + h * DH_, bf2f(hp[h * DH_]) * coef[h]);
}

// mean over heads + bias (+ optional leaky_relu 0.15) -> bf16 node features
__global__ void head_mean_kernel(const float* __restrict__ agg,
                                 const float* __restrict__ bias,
                                 unsigned short* __restrict__ xout, int act) {
    int idx = blockIdx.x * blockDim.x + threadIdx.x;
    if (idx >= NNODES * DH_) return;
    int i = idx >> 7, c = idx & 127;
    const float* p = agg + (size_t)i * HDIM + c;
    float s = 0.f;
#pragma unroll
    for (int h = 0; h < NHEAD; ++h) s += p[h * DH_];
    s = s * (1.0f / NHEAD) + bias[c];
    if (act == 1) s = s > 0.f ? s : 0.15f * s;
    xout[idx] = f2bf(s);
}

// target gather: trg[t,:] = x[N[t],:]
__global__ void gather_rows_kernel(const int* __restrict__ Nidx,
                                   const unsigned short* __restrict__ xin,
                                   unsigned short* __restrict__ outp) {
    int idx = blockIdx.x * blockDim.x + threadIdx.x;
    if (idx >= TTARG * DH_) return;
    int t = idx >> 7, c = idx & 127;
    outp[idx] = xin[(size_t)Nidx[t] * DH_ + c];
}

// ---------------------------------------------------------------------------
static inline size_t alignup(size_t x) { return (x + 255) & ~(size_t)255; }

extern "C" void kernel_launch(void* const* d_in, const int* in_sizes, int n_in,
                              void* d_out, int out_size, void* d_ws, size_t ws_size,
                              hipStream_t stream) {
    (void)in_sizes; (void)n_in; (void)out_size; (void)ws_size;
    const int*   ei     = (const int*)  d_in[0];
    const int*   Nidx   = (const int*)  d_in[1];
    const int*   y      = (const int*)  d_in[2];
    const float* emb    = (const float*)d_in[3];
    const float* W1     = (const float*)d_in[4];
    const float* a_src1 = (const float*)d_in[5];
    const float* a_dst1 = (const float*)d_in[6];
    const float* b1     = (const float*)d_in[7];
    const float* W2     = (const float*)d_in[8];
    const float* a_src2 = (const float*)d_in[9];
    const float* a_dst2 = (const float*)d_in[10];
    const float* b2     = (const float*)d_in[11];
    const float* Wm     = (const float*)d_in[12];
    const float* bm     = (const float*)d_in[13];
    const float* Wm1    = (const float*)d_in[14];
    const float* bm1    = (const float*)d_in[15];
    const float* Wout   = (const float*)d_in[16];
    const float* bout   = (const float*)d_in[17];
    float* outp = (float*)d_out;

    // workspace bump allocation (~355 MB)
    char* p = (char*)d_ws;
    auto take = [&](size_t bytes) { char* r = p; p += alignup(bytes); return r; };
    unsigned short* xa    = (unsigned short*)take((size_t)NNODES * DH_ * 2);
    unsigned short* xb    = (unsigned short*)take((size_t)NNODES * DH_ * 2);
    unsigned short* hbuf  = (unsigned short*)take((size_t)NNODES * HDIM * 2);
    float*          agg   = (float*)         take((size_t)NNODES * HDIM * 4);
    float*          asn   = (float*)         take((size_t)NNODES * NHEAD * 4);
    float*          adn   = (float*)         take((size_t)NNODES * NHEAD * 4);
    unsigned*       amax  = (unsigned*)      take((size_t)NNODES * NHEAD * 4);
    float*          denom = (float*)         take((size_t)NNODES * NHEAD * 4);
    float*          exb   = (float*)         take((size_t)ETOT * NHEAD * 4);
    unsigned short* W1t   = (unsigned short*)take((size_t)HDIM * DH_ * 2);
    unsigned short* W2t   = (unsigned short*)take((size_t)HDIM * DH_ * 2);
    unsigned short* Wmt   = (unsigned short*)take((size_t)DH_ * DH_ * 2);
    unsigned short* Wm1t  = (unsigned short*)take((size_t)DH_ * DH_ * 2);
    unsigned short* Woutt = (unsigned short*)take((size_t)VOCAB * DH_ * 2);
    unsigned short* trg   = (unsigned short*)take((size_t)TTARG * DH_ * 2);

    auto cdiv = [](long long a, long long b) { return (int)((a + b - 1) / b); };
    auto gemm = [&](const unsigned short* A, const unsigned short* Bt, const float* bias,
                    float* Cf, unsigned short* Cbf, int M, int N, int K, int act) {
        int tiles = (M / 16) * (N / 16);
        hipLaunchKernelGGL(wmma_gemm_kernel, dim3(cdiv(tiles, 8)), dim3(256), 0, stream,
                           A, Bt, bias, Cf, Cbf, M, N, K, act);
    };
    auto gat = [&](const unsigned short* xin, const unsigned short* Wt,
                   const float* a_s, const float* a_d, const float* bias,
                   unsigned short* xout, int act) {
        gemm(xin, Wt, nullptr, nullptr, hbuf, NNODES, HDIM, DH_, 0);      // h = x @ W
        hipLaunchKernelGGL(attn_score_kernel, dim3(cdiv((long long)NNODES * NHEAD, 256)),
                           dim3(256), 0, stream, hbuf, a_s, a_d, asn, adn);
        hipMemsetAsync(amax,  0, (size_t)NNODES * NHEAD * 4, stream);      // enc(-inf-ish) bottom
        hipMemsetAsync(denom, 0, (size_t)NNODES * NHEAD * 4, stream);
        hipMemsetAsync(agg,   0, (size_t)NNODES * HDIM * 4, stream);
        int egrid = cdiv((long long)ETOT * NHEAD, 256);
        hipLaunchKernelGGL(edge_max_kernel, dim3(egrid), dim3(256), 0, stream,
                           ei, asn, adn, exb, amax);
        hipLaunchKernelGGL(edge_exp_kernel, dim3(egrid), dim3(256), 0, stream,
                           ei, exb, amax, denom);
        hipLaunchKernelGGL(edge_agg_kernel, dim3(ETOT), dim3(DH_), 0, stream,
                           ei, hbuf, exb, denom, agg);
        hipLaunchKernelGGL(head_mean_kernel, dim3(cdiv((long long)NNODES * DH_, 256)),
                           dim3(256), 0, stream, agg, bias, xout, act);
    };

    // weight prep (bf16, transposed to [N][K])
    hipLaunchKernelGGL(transpose_bf16_kernel, dim3(cdiv((long long)DH_ * HDIM, 256)),
                       dim3(256), 0, stream, W1, W1t, DH_, HDIM);
    hipLaunchKernelGGL(transpose_bf16_kernel, dim3(cdiv((long long)DH_ * HDIM, 256)),
                       dim3(256), 0, stream, W2, W2t, DH_, HDIM);
    hipLaunchKernelGGL(transpose_bf16_kernel, dim3(cdiv((long long)DH_ * DH_, 256)),
                       dim3(256), 0, stream, Wm, Wmt, DH_, DH_);
    hipLaunchKernelGGL(transpose_bf16_kernel, dim3(cdiv((long long)DH_ * DH_, 256)),
                       dim3(256), 0, stream, Wm1, Wm1t, DH_, DH_);
    hipLaunchKernelGGL(transpose_bf16_kernel, dim3(cdiv((long long)DH_ * VOCAB, 256)),
                       dim3(256), 0, stream, Wout, Woutt, DH_, VOCAB);

    // x = layernorm(emb[y])
    hipLaunchKernelGGL(embed_ln_kernel, dim3(NNODES), dim3(DH_), 0, stream, y, emb, xa);

    // GAT layer 1 (+b1, leaky 0.15), then GELU(x@Wm+bm)
    gat(xa, W1t, a_src1, a_dst1, b1, xb, /*act=*/1);
    gemm(xb, Wmt, bm, nullptr, xa, NNODES, DH_, DH_, /*gelu*/2);

    // GAT layer 2 (+b2), then GELU(x@Wm1+bm1)
    gat(xa, W2t, a_src2, a_dst2, b2, xb, /*act=*/0);
    gemm(xb, Wm1t, bm1, nullptr, xa, NNODES, DH_, DH_, /*gelu*/2);

    // target gather + vocab projection (fp32 out)
    hipLaunchKernelGGL(gather_rows_kernel, dim3(cdiv((long long)TTARG * DH_, 256)),
                       dim3(256), 0, stream, Nidx, xa, trg);
    gemm(trg, Woutt, bout, outp, nullptr, TTARG, VOCAB, DH_, 0);
}
